// Decoder_7799660610145
// MI455X (gfx1250) — compile-verified, matrix operand order checked by reading, not run on hardware
//
#include <hip/hip_runtime.h>
#include <hip/hip_bf16.h>
#include <math.h>

// ---------------------------------------------------------------------------
// Tacotron2 decoder for MI455X (gfx1250, wave32, WMMA bf16 16x16x32).
//
//  * One-time: pack all weights to bf16 WMMA-fragment tiles; precompute
//    prenet (two WMMA GEMMs over all 800*32 rows) and pmem = enc@memory_W.
//  * Per step (800 sequential iters): attn-LSTM gate GEMM (WMMA, 2 acc/wave),
//    pointwise LSTM, attention (VALU conv+softmax+context), dec-LSTM GEMM,
//    pointwise, projection.
//  * bf16 weights (~36MB) stay L2-resident (192MB) across all steps.
//  * Each wave: per K-tile = 1 A-frag load + 2 B-frag loads + 2 independent
//    v_wmma (pipelined), + global_prefetch_b8 of next K-tile.
// ---------------------------------------------------------------------------

#define B_SZ   32
#define T_ENCC 256
#define E_DIM  512
#define P_DIM  256
#define N_MELC 80
#define T_DECC 800
#define A_DIMC 128
#define N_FILT 32
#define K_CONV 31
#define PAD_C  15

typedef __attribute__((ext_vector_type(16))) __bf16 v16bf;
typedef __attribute__((ext_vector_type(8)))  float  v8f;

static __device__ __forceinline__ float sigmoidf_(float x) {
    return 1.0f / (1.0f + __expf(-x));
}

static __device__ __forceinline__ unsigned drop_hash(unsigned r, unsigned c, unsigned seed) {
    unsigned h = (r * 2654435761u) ^ (c * 40503u) ^ seed;
    h = h * 747796405u + 2891336453u;
    h ^= h >> 17;
    h *= 0xed5ad4bbu;
    h ^= h >> 11;
    return h & 1u;
}

// ---------------------------------------------------------------------------
// Weight packer: WMMA B-fragment tile layout.
// Tile (kt,nt) of 32(K)x16(N) -> 512 bf16, lane l holds 16 contiguous halfs:
//   n_in_tile = l & 15 ; k_in_tile = (l>>4)*16 + j
// nmajor==1: src is [N,K] row-major; nmajor==0: src is [K,N] row-major.
// ---------------------------------------------------------------------------
__global__ void pack_b_kernel(__bf16* __restrict__ dst, int ntilesTot,
                              const float* __restrict__ src, int src_ld,
                              int src_coloff, int dst_koff, int klen, int N,
                              int nmajor) {
    long long idx = (long long)blockIdx.x * 256 + threadIdx.x;
    long long total = (long long)N * klen;
    if (idx >= total) return;
    int n  = (int)(idx / klen);
    int kr = (int)(idx % klen);
    int k  = dst_koff + kr;
    int kt = k >> 5, kin = k & 31;
    int nt = n >> 4, nin = n & 15;
    int l  = nin | ((kin >> 4) << 4);
    int j  = kin & 15;
    float v;
    if (nmajor) v = src[(size_t)n * src_ld + src_coloff + kr];
    else        v = src[(size_t)(src_coloff + kr) * src_ld + n];
    dst[((size_t)kt * ntilesTot + nt) * 512 + l * 16 + j] = (__bf16)v;
}

__global__ void combine_bias_kernel(float* __restrict__ dst,
                                    const float* __restrict__ b1,
                                    const float* __restrict__ b2, int n) {
    int i = blockIdx.x * 256 + threadIdx.x;
    if (i < n) dst[i] = b1[i] + b2[i];
}

__global__ void f32_to_bf16_kernel(__bf16* __restrict__ dst,
                                   const float* __restrict__ src, long long n) {
    long long i = (long long)blockIdx.x * 256 + threadIdx.x;
    if (i < n) dst[i] = (__bf16)src[i];
}

// Shifted decoder input: xs[t][b][m] (m padded 80->96), t=0 is go-frame.
__global__ void build_xshift_kernel(__bf16* __restrict__ dst,
                                    const float* __restrict__ dec_in) {
    long long idx = (long long)blockIdx.x * 256 + threadIdx.x;
    long long total = (long long)T_DECC * B_SZ * 96;
    if (idx >= total) return;
    int m = (int)(idx % 96);
    int r = (int)(idx / 96);
    int b = r % B_SZ;
    int t = r / B_SZ;
    float v = 0.0f;
    if (m < N_MELC && t > 0)
        v = dec_in[(size_t)b * N_MELC * T_DECC + (size_t)m * T_DECC + (t - 1)];
    dst[idx] = (__bf16)v;
}

// ---------------------------------------------------------------------------
// Generic WMMA bf16 GEMM: C[M,N] = A[M,K] * Bpacked[K,N].
// 256 thr = 8 waves; wave -> 16 rows x 2 N-tiles (2 accumulators).
// Block covers 32 rows x 128 cols.  grid = (ceil(N/128), M/32).
// mode 0: f32 to outF ; mode 1: relu->dropout(0.5,*2,hash)->bf16 to outB.
// ---------------------------------------------------------------------------
__global__ void wmma_gemm_kernel(const __bf16* __restrict__ A, int lda,
                                 const __bf16* __restrict__ Bp, int ktiles,
                                 int ntilesTot, int N,
                                 float* __restrict__ outF,
                                 __bf16* __restrict__ outB,
                                 int mode, unsigned seed) {
    int w    = threadIdx.x >> 5;
    int lane = threadIdx.x & 31;
    int mtile  = (w & 1);
    int ntile0 = (blockIdx.x * 4 + (w >> 1)) * 2;
    if (ntile0 >= ntilesTot) return;  // uniform per wave (ntilesTot even)
    int row  = blockIdx.y * 32 + mtile * 16 + (lane & 15);
    int kofs = (lane >> 4) * 16;

    v8f acc0 = {}, acc1 = {};
    for (int kt = 0; kt < ktiles; ++kt) {
        v16bf a  = *(const v16bf*)(A + (size_t)row * lda + kt * 32 + kofs);
        const __bf16* bbase = Bp + ((size_t)kt * ntilesTot + ntile0) * 512 + lane * 16;
        v16bf b0 = *(const v16bf*)bbase;
        v16bf b1 = *(const v16bf*)(bbase + 512);
        if (kt + 1 < ktiles)
            __builtin_prefetch(bbase + (size_t)ntilesTot * 512, 0, 3);
        acc0 = __builtin_amdgcn_wmma_f32_16x16x32_bf16(
            false, a, false, b0, (short)0, acc0, false, false);
        acc1 = __builtin_amdgcn_wmma_f32_16x16x32_bf16(
            false, a, false, b1, (short)0, acc1, false, false);
    }

    int n     = lane & 15;
    int mbase = (lane < 16) ? 0 : 8;
#pragma unroll
    for (int r = 0; r < 8; ++r) {
        int m = blockIdx.y * 32 + mtile * 16 + mbase + r;
#pragma unroll
        for (int u = 0; u < 2; ++u) {
            int col = (ntile0 + u) * 16 + n;
            float v = (u == 0) ? acc0[r] : acc1[r];
            if (mode == 0) {
                outF[(size_t)m * N + col] = v;
            } else {
                v = fmaxf(v, 0.0f);
                v = drop_hash((unsigned)m, (unsigned)col, seed) ? v * 2.0f : 0.0f;
                outB[(size_t)m * N + col] = (__bf16)v;
            }
        }
    }
}

// ---------------------------------------------------------------------------
// LSTM gate GEMM: gates[32,4096] = concat(s0,s1,s2)[32,K] * Bpacked + bias.
// A read directly from the three bf16 state buffers (concat boundaries are
// multiples of K-tile 32).  Wave owns 16 rows x 2 N-tiles (2 accumulators):
// per K-tile 1 A load + 2 B loads + 2 independent v_wmma.
// grid = 32 blocks x 256 thr (block = 32 rows x 128 cols of 32x4096).
// ---------------------------------------------------------------------------
__global__ void lstm_gemm_kernel(const __bf16* __restrict__ s0, int ld0, int k1,
                                 const __bf16* __restrict__ s1, int ld1, int k2,
                                 const __bf16* __restrict__ s2, int ld2,
                                 const __bf16* __restrict__ Bp, int ktiles,
                                 const float* __restrict__ bias,
                                 float* __restrict__ gates) {
    int w    = threadIdx.x >> 5;
    int lane = threadIdx.x & 31;
    int mtile  = (w & 1);
    int ntile0 = (blockIdx.x * 4 + (w >> 1)) * 2;  // 0..254 (N=4096 -> 256 tiles)
    int row  = mtile * 16 + (lane & 15);
    int kofs = (lane >> 4) * 16;

    v8f acc0 = {}, acc1 = {};
    for (int kt = 0; kt < ktiles; ++kt) {
        int k = kt * 32 + kofs;
        const __bf16* sp; int ld, kk;
        if (k < k1)      { sp = s0; ld = ld0; kk = k; }
        else if (k < k2) { sp = s1; ld = ld1; kk = k - k1; }
        else             { sp = s2; ld = ld2; kk = k - k2; }
        v16bf a  = *(const v16bf*)(sp + (size_t)row * ld + kk);
        const __bf16* bbase = Bp + ((size_t)kt * 256 + ntile0) * 512 + lane * 16;
        v16bf b0 = *(const v16bf*)bbase;
        v16bf b1 = *(const v16bf*)(bbase + 512);
        if (kt + 1 < ktiles)
            __builtin_prefetch(bbase + (size_t)256 * 512, 0, 3);
        acc0 = __builtin_amdgcn_wmma_f32_16x16x32_bf16(
            false, a, false, b0, (short)0, acc0, false, false);
        acc1 = __builtin_amdgcn_wmma_f32_16x16x32_bf16(
            false, a, false, b1, (short)0, acc1, false, false);
    }

    int n     = lane & 15;
    int mbase = (lane < 16) ? 0 : 8;
#pragma unroll
    for (int r = 0; r < 8; ++r) {
        int m = mtile * 16 + mbase + r;
#pragma unroll
        for (int u = 0; u < 2; ++u) {
            int col = (ntile0 + u) * 16 + n;
            float v = (u == 0) ? acc0[r] : acc1[r];
            gates[(size_t)m * 4096 + col] = v + bias[col];
        }
    }
}

// LSTM pointwise: gates [32,4096] order (i,f,g,o); updates c,h; writes hbf.
__global__ void lstm_act_kernel(const float* __restrict__ gates,
                                float* __restrict__ c, float* __restrict__ h,
                                __bf16* __restrict__ hbf) {
    int idx = blockIdx.x * 256 + threadIdx.x;
    if (idx >= B_SZ * 1024) return;
    int b = idx >> 10, j = idx & 1023;
    const float* g = gates + (size_t)b * 4096;
    float gi = sigmoidf_(g[j]);
    float gf = sigmoidf_(g[1024 + j]);
    float gg = tanhf(g[2048 + j]);
    float go = sigmoidf_(g[3072 + j]);
    float cc = gf * c[idx] + gi * gg;
    c[idx] = cc;
    float hh = go * tanhf(cc);
    h[idx] = hh;
    hbf[idx] = (__bf16)hh;
}

// ---------------------------------------------------------------------------
// Location-sensitive attention. One block per batch row, 256 threads.
// ---------------------------------------------------------------------------
__global__ void attention_kernel(int t,
                                 const float* __restrict__ ah,
                                 float* __restrict__ aw, float* __restrict__ awc,
                                 const float* __restrict__ pmem,
                                 const float* __restrict__ enc,
                                 const float* __restrict__ qW,
                                 const float* __restrict__ vw,
                                 const float* __restrict__ lcW,
                                 const float* __restrict__ llW,
                                 const int* __restrict__ in_len,
                                 float* __restrict__ ctx, __bf16* __restrict__ ctxbf,
                                 float* __restrict__ align_out) {
    int b = blockIdx.x;
    int tid = threadIdx.x;
    __shared__ float s_ah[1024];
    __shared__ float s_q[128];
    __shared__ float s_aw[T_ENCC + 30];
    __shared__ float s_awc[T_ENCC + 30];
    __shared__ float s_e[T_ENCC];
    __shared__ float s_red[256];

    for (int i = tid; i < 1024; i += 256) s_ah[i] = ah[(size_t)b * 1024 + i];
    for (int i = tid; i < T_ENCC + 30; i += 256) {
        int p = i - PAD_C;
        float va = 0.f, vc = 0.f;
        if (p >= 0 && p < T_ENCC) { va = aw[b * T_ENCC + p]; vc = awc[b * T_ENCC + p]; }
        s_aw[i] = va; s_awc[i] = vc;
    }
    __syncthreads();

    if (tid < 128) {
        float s = 0.f;
        for (int k = 0; k < 1024; ++k) s += s_ah[k] * qW[k * 128 + tid];
        s_q[tid] = s;
    }
    __syncthreads();

    int pos = tid;
    float lf[N_FILT];
#pragma unroll 4
    for (int f = 0; f < N_FILT; ++f) {
        float s = 0.f;
        const float* wf = lcW + f * (2 * K_CONV);
        for (int kk = 0; kk < K_CONV; ++kk)
            s += s_aw[pos + kk] * wf[kk] + s_awc[pos + kk] * wf[K_CONV + kk];
        lf[f] = s;
    }
    float e = 0.f;
    const float* pm = pmem + ((size_t)b * T_ENCC + pos) * A_DIMC;
    for (int a = 0; a < A_DIMC; ++a) {
        float s = s_q[a] + pm[a];
        for (int f = 0; f < N_FILT; ++f) s += lf[f] * llW[f * A_DIMC + a];
        e += tanhf(s) * vw[a];
    }
    if (pos >= in_len[b]) e = -1e9f;
    s_e[pos] = e;

    // softmax over 256
    s_red[tid] = e;
    __syncthreads();
    for (int s = 128; s > 0; s >>= 1) {
        if (tid < s) s_red[tid] = fmaxf(s_red[tid], s_red[tid + s]);
        __syncthreads();
    }
    float mx = s_red[0];
    __syncthreads();
    float ex = __expf(e - mx);
    s_red[tid] = ex;
    __syncthreads();
    for (int s = 128; s > 0; s >>= 1) {
        if (tid < s) s_red[tid] += s_red[tid + s];
        __syncthreads();
    }
    float p = ex / s_red[0];
    aw[b * T_ENCC + pos] = p;
    awc[b * T_ENCC + pos] = s_awc[pos + PAD_C] + p;
    align_out[((size_t)b * T_DECC + t) * T_ENCC + pos] = p;
    s_e[pos] = p;
    __syncthreads();

    for (int ecol = tid; ecol < E_DIM; ecol += 256) {
        float s = 0.f;
        for (int q = 0; q < T_ENCC; ++q)
            s += s_e[q] * enc[((size_t)b * T_ENCC + q) * E_DIM + ecol];
        ctx[b * E_DIM + ecol] = s;
        ctxbf[b * E_DIM + ecol] = (__bf16)s;
    }
}

// Projection + gate, writes directly into output layout.
__global__ void proj_kernel(int t,
                            const float* __restrict__ dh, const float* __restrict__ ctx,
                            const float* __restrict__ pW, const float* __restrict__ pb,
                            const float* __restrict__ gW, const float* __restrict__ gb,
                            float* __restrict__ out, long long gate_off) {
    int idx = blockIdx.x * 256 + threadIdx.x;
    if (idx < B_SZ * N_MELC) {
        int b = idx / N_MELC, j = idx % N_MELC;
        float s = pb[j];
        const float* dhb = dh + (size_t)b * 1024;
        const float* cxb = ctx + (size_t)b * E_DIM;
        for (int k = 0; k < 1024; ++k) s += dhb[k] * pW[(size_t)k * N_MELC + j];
        for (int k = 0; k < E_DIM; ++k) s += cxb[k] * pW[(size_t)(1024 + k) * N_MELC + j];
        out[((size_t)b * N_MELC + j) * T_DECC + t] = s;
    } else if (idx < B_SZ * N_MELC + B_SZ) {
        int b = idx - B_SZ * N_MELC;
        float s = gb[0];
        const float* dhb = dh + (size_t)b * 1024;
        const float* cxb = ctx + (size_t)b * E_DIM;
        for (int k = 0; k < 1024; ++k) s += dhb[k] * gW[k];
        for (int k = 0; k < E_DIM; ++k) s += cxb[k] * gW[1024 + k];
        out[gate_off + (size_t)b * T_DECC + t] = s;
    }
}

// ---------------------------------------------------------------------------
// Host launcher
// ---------------------------------------------------------------------------
extern "C" void kernel_launch(void* const* d_in, const int* in_sizes, int n_in,
                              void* d_out, int out_size, void* d_ws, size_t ws_size,
                              hipStream_t stream) {
    const float* enc    = (const float*)d_in[0];
    const float* dec_in = (const float*)d_in[1];
    const int*   in_len = (const int*)d_in[2];
    const float* w1     = (const float*)d_in[3];
    const float* w2     = (const float*)d_in[4];
    const float* aWih   = (const float*)d_in[5];
    const float* aWhh   = (const float*)d_in[6];
    const float* a_bih  = (const float*)d_in[7];
    const float* a_bhh  = (const float*)d_in[8];
    const float* qW     = (const float*)d_in[9];
    const float* mW     = (const float*)d_in[10];
    const float* vw     = (const float*)d_in[11];
    const float* lcW    = (const float*)d_in[12];
    const float* llW    = (const float*)d_in[13];
    const float* dWih   = (const float*)d_in[14];
    const float* dWhh   = (const float*)d_in[15];
    const float* d_bih  = (const float*)d_in[16];
    const float* d_bhh  = (const float*)d_in[17];
    const float* pW     = (const float*)d_in[18];
    const float* pb     = (const float*)d_in[19];
    const float* gW     = (const float*)d_in[20];
    const float* gb     = (const float*)d_in[21];
    float* out = (float*)d_out;

    const long long GATE_OFF  = (long long)B_SZ * N_MELC * T_DECC;       // 2,048,000
    const long long ALIGN_OFF = GATE_OFF + (long long)B_SZ * T_DECC;     // 2,073,600

    size_t off = 0;
    auto carve = [&](size_t bytes) -> char* {
        char* p = (char*)d_ws + off;
        off += (bytes + 255) & ~(size_t)255;
        return p;
    };
    __bf16* B1p   = (__bf16*)carve((size_t)1792 * 4096 * 2);
    __bf16* B2p   = (__bf16*)carve((size_t)2560 * 4096 * 2);
    __bf16* W1p   = (__bf16*)carve((size_t)96 * 256 * 2);
    __bf16* W2p   = (__bf16*)carve((size_t)256 * 256 * 2);
    __bf16* WMp   = (__bf16*)carve((size_t)512 * 128 * 2);
    __bf16* encb  = (__bf16*)carve((size_t)B_SZ * T_ENCC * E_DIM * 2);
    float*  pmem  = (float*) carve((size_t)B_SZ * T_ENCC * A_DIMC * 4);
    __bf16* xs    = (__bf16*)carve((size_t)T_DECC * B_SZ * 96 * 2);
    __bf16* h1    = (__bf16*)carve((size_t)T_DECC * B_SZ * 256 * 2);
    __bf16* pre   = (__bf16*)carve((size_t)T_DECC * B_SZ * 256 * 2);
    float*  gates1= (float*) carve((size_t)B_SZ * 4096 * 4);
    float*  gates2= (float*) carve((size_t)B_SZ * 4096 * 4);
    float*  bias1 = (float*) carve(4096 * 4);
    float*  bias2 = (float*) carve(4096 * 4);
    float*  ah    = (float*) carve((size_t)B_SZ * 1024 * 4);
    float*  ac    = (float*) carve((size_t)B_SZ * 1024 * 4);
    float*  dh    = (float*) carve((size_t)B_SZ * 1024 * 4);
    float*  dc    = (float*) carve((size_t)B_SZ * 1024 * 4);
    __bf16* ahbf  = (__bf16*)carve((size_t)B_SZ * 1024 * 2);
    __bf16* dhbf  = (__bf16*)carve((size_t)B_SZ * 1024 * 2);
    float*  ctx   = (float*) carve((size_t)B_SZ * E_DIM * 4);
    __bf16* ctxbf = (__bf16*)carve((size_t)B_SZ * E_DIM * 2);
    float*  aw    = (float*) carve((size_t)B_SZ * T_ENCC * 4);
    float*  awc   = (float*) carve((size_t)B_SZ * T_ENCC * 4);

    hipMemsetAsync(ah,    0, (size_t)B_SZ * 1024 * 4, stream);
    hipMemsetAsync(ac,    0, (size_t)B_SZ * 1024 * 4, stream);
    hipMemsetAsync(dh,    0, (size_t)B_SZ * 1024 * 4, stream);
    hipMemsetAsync(dc,    0, (size_t)B_SZ * 1024 * 4, stream);
    hipMemsetAsync(ahbf,  0, (size_t)B_SZ * 1024 * 2, stream);
    hipMemsetAsync(dhbf,  0, (size_t)B_SZ * 1024 * 2, stream);
    hipMemsetAsync(ctx,   0, (size_t)B_SZ * E_DIM * 4, stream);
    hipMemsetAsync(ctxbf, 0, (size_t)B_SZ * E_DIM * 2, stream);
    hipMemsetAsync(aw,    0, (size_t)B_SZ * T_ENCC * 4, stream);
    hipMemsetAsync(awc,   0, (size_t)B_SZ * T_ENCC * 4, stream);
    hipMemsetAsync(W1p,   0, (size_t)96 * 256 * 2, stream);

    auto gblk = [](long long n) { return (unsigned)((n + 255) / 256); };

    // ---- weight packing (bf16, WMMA tile layout) ----
    pack_b_kernel<<<gblk((long long)4096 * 768),  256, 0, stream>>>(B1p, 256, aWih, 768,  0,    0,    768,  4096, 1);
    pack_b_kernel<<<gblk((long long)4096 * 1024), 256, 0, stream>>>(B1p, 256, aWhh, 1024, 0,    768,  1024, 4096, 1);
    pack_b_kernel<<<gblk((long long)4096 * 1024), 256, 0, stream>>>(B2p, 256, dWih, 1536, 0,    0,    1024, 4096, 1);
    pack_b_kernel<<<gblk((long long)4096 * 512),  256, 0, stream>>>(B2p, 256, dWih, 1536, 1024, 1024, 512,  4096, 1);
    pack_b_kernel<<<gblk((long long)4096 * 1024), 256, 0, stream>>>(B2p, 256, dWhh, 1024, 0,    1536, 1024, 4096, 1);
    pack_b_kernel<<<gblk((long long)256 * 80),  256, 0, stream>>>(W1p, 16, w1, 80,  0, 0, 80,  256, 1);
    pack_b_kernel<<<gblk((long long)256 * 256), 256, 0, stream>>>(W2p, 16, w2, 256, 0, 0, 256, 256, 1);
    pack_b_kernel<<<gblk((long long)128 * 512), 256, 0, stream>>>(WMp, 8, mW, 128, 0, 0, 512, 128, 0);

    combine_bias_kernel<<<16, 256, 0, stream>>>(bias1, a_bih, a_bhh, 4096);
    combine_bias_kernel<<<16, 256, 0, stream>>>(bias2, d_bih, d_bhh, 4096);

    // ---- one-time precompute ----
    f32_to_bf16_kernel<<<gblk((long long)B_SZ * T_ENCC * E_DIM), 256, 0, stream>>>(
        encb, enc, (long long)B_SZ * T_ENCC * E_DIM);
    build_xshift_kernel<<<gblk((long long)T_DECC * B_SZ * 96), 256, 0, stream>>>(xs, dec_in);

    // pmem = enc_bf16 [8192,512] @ memory_W -> f32 [8192,128]  (8 N-tiles)
    wmma_gemm_kernel<<<dim3(1, 8192 / 32), 256, 0, stream>>>(
        encb, 512, WMp, 16, 8, 128, pmem, nullptr, 0, 0u);
    // prenet layer 1: [25600,96] @ [96,256] -> relu/dropout -> bf16 h1
    wmma_gemm_kernel<<<dim3(2, 25600 / 32), 256, 0, stream>>>(
        xs, 96, W1p, 3, 16, 256, nullptr, h1, 1, 0x13572468u);
    // prenet layer 2: [25600,256] @ [256,256] -> relu/dropout -> bf16 pre
    wmma_gemm_kernel<<<dim3(2, 25600 / 32), 256, 0, stream>>>(
        h1, 256, W2p, 8, 16, 256, nullptr, pre, 1, 0x9e3779b9u);

    // ---- sequential decode: 800 steps ----
    for (int t = 0; t < T_DECC; ++t) {
        const __bf16* pre_t = pre + (size_t)t * B_SZ * 256;
        // attn-LSTM gates: A = [pre_t(256) | ctx(512) | ah(1024)], K=1792
        lstm_gemm_kernel<<<32, 256, 0, stream>>>(
            pre_t, 256, 256, ctxbf, 512, 768, ahbf, 1024,
            B1p, 56, bias1, gates1);
        lstm_act_kernel<<<(B_SZ * 1024 + 255) / 256, 256, 0, stream>>>(gates1, ac, ah, ahbf);

        attention_kernel<<<B_SZ, 256, 0, stream>>>(
            t, ah, aw, awc, pmem, enc, qW, vw, lcW, llW, in_len,
            ctx, ctxbf, out + ALIGN_OFF);

        // dec-LSTM gates: A = [ah(1024) | ctx(512) | dh(1024)], K=2560
        lstm_gemm_kernel<<<32, 256, 0, stream>>>(
            ahbf, 1024, 1024, ctxbf, 512, 1536, dhbf, 1024,
            B2p, 80, bias2, gates2);
        lstm_act_kernel<<<(B_SZ * 1024 + 255) / 256, 256, 0, stream>>>(gates2, dc, dh, dhbf);

        proj_kernel<<<(B_SZ * N_MELC + B_SZ + 255) / 256, 256, 0, stream>>>(
            t, dh, ctx, pW, pb, gW, gb, out, GATE_OFF);
    }
}
